// LSTM_23441931501830
// MI455X (gfx1250) — matrix operand order repaired and checked by reading, not compiled, not run on hardware
//
#include <hip/hip_runtime.h>
#include <hip/hip_bf16.h>

typedef __attribute__((ext_vector_type(4)))  float    v4f;
typedef __attribute__((ext_vector_type(8)))  float    v8f;
typedef __attribute__((ext_vector_type(16))) _Float16 v16h;

#define BATCH 4096
#define SEQT  512
#define XPITCH 65   // [t][row] pitch: 16 batch columns -> 16 distinct LDS banks

// Gate-row scale: i/f/o rows (0..63, 96..127) pre-scaled by 0.5 so that
// sigmoid(z) = 0.5*tanh(0.5*z)+0.5 needs no pre-multiply after the WMMA.
__device__ __host__ __forceinline__ float gate_scale(int row) {
  return (row >= 64 && row < 96) ? 1.0f : 0.5f;
}

// ---------------- small helpers ----------------

__device__ __forceinline__ v8f zero8() {
  v8f r;
#pragma unroll
  for (int i = 0; i < 8; ++i) r[i] = 0.0f;
  return r;
}

__device__ __forceinline__ v16h zeroh() {
  v16h r;
#pragma unroll
  for (int i = 0; i < 16; ++i) r[i] = (_Float16)0.0f;
  return r;
}

// Native v_tanh_f32 on CDNA5.
__device__ __forceinline__ float ftanh_(float x) {
#if __has_builtin(__builtin_amdgcn_tanhf)
  return __builtin_amdgcn_tanhf(x);
#else
  return 2.0f * __builtin_amdgcn_rcpf(1.0f + __expf(-2.0f * x)) - 1.0f;
#endif
}
// sigmoid of the ORIGINAL gate value z, given the pre-scaled input x = 0.5*z
__device__ __forceinline__ float fsig_pre(float x) {
#if __has_builtin(__builtin_amdgcn_tanhf)
  return fmaf(0.5f, __builtin_amdgcn_tanhf(x), 0.5f);
#else
  return __builtin_amdgcn_rcpf(1.0f + __expf(-2.0f * x));
#endif
}

__device__ __forceinline__ v8f sig8p(v8f x) {   // input pre-scaled by 0.5
  v8f r;
#pragma unroll
  for (int i = 0; i < 8; ++i) r[i] = fsig_pre(x[i]);
  return r;
}
__device__ __forceinline__ v8f tanh8(v8f x) {
  v8f r;
#pragma unroll
  for (int i = 0; i < 8; ++i) r[i] = ftanh_(x[i]);
  return r;
}

// r = a*s + b, elementwise with scalar s
__device__ __forceinline__ v8f fma8s(v8f a, float s, v8f b) {
  v8f r;
#pragma unroll
  for (int i = 0; i < 8; ++i) r[i] = fmaf(s, a[i], b[i]);
  return r;
}

// load 8 consecutive floats (two float4) from an LDS-backed v4f array
__device__ __forceinline__ v8f ld8(const v4f* p, int blk, int hoff) {
  v4f a = p[4 * blk + hoff];
  v4f b = p[4 * blk + hoff + 1];
  v8f r;
#pragma unroll
  for (int i = 0; i < 4; ++i) { r[i] = a[i]; r[4 + i] = b[i]; }
  return r;
}

// pack two f32 into one 32-bit word of packed f16 (v_cvt_pk_rtz_f16_f32)
__device__ __forceinline__ unsigned cvtpk(float a, float b) {
  auto p = __builtin_amdgcn_cvt_pkrtz(a, b);   // __fp16 ext_vector_type(2)
  return __builtin_bit_cast(unsigned, p);
}

__device__ __forceinline__ v8f wmma16(v16h a, v16h b, v8f c) {
  // D = A(16x32 f16) * B(32x16 f16) + C(16x16 f32)
  return __builtin_amdgcn_wmma_f32_16x16x32_f16(
      /*neg_a=*/false, a, /*neg_b=*/false, b,
      /*c_mod=*/(short)0, c, /*reuse_a=*/false, /*reuse_b=*/false);
}

// Convert new h (two 16x16 f32 D-layout blocks) into the 32x16 f16 B-operand
// layout. Convert to packed f16 FIRST, then xor-16 shuffle + select on the
// packed 32-bit words: 8 cvt_pk + 8 shfl + 8 cndmask total.
__device__ __forceinline__ v16h packB(v8f b0, v8f b1, bool lo) {
  unsigned p0[4], p1[4];
#pragma unroll
  for (int j = 0; j < 4; ++j) {
    p0[j] = cvtpk(b0[2 * j], b0[2 * j + 1]);
    p1[j] = cvtpk(b1[2 * j], b1[2 * j + 1]);
  }
  union { v16h v; unsigned u[8]; } r;
#pragma unroll
  for (int j = 0; j < 4; ++j) {
    unsigned s0 = (unsigned)__shfl_xor((int)p0[j], 16, 32);
    unsigned s1 = (unsigned)__shfl_xor((int)p1[j], 16, 32);
    r.u[j]     = lo ? p0[j] : s1;   // K offsets 0..7  within lane
    r.u[4 + j] = lo ? s0 : p1[j];   // K offsets 8..15 within lane
  }
  return r.v;
}

// Stage a [128,32] row-major f32 weight matrix into the per-lane f16
// A-operand blobs for 8 (16x32) blocks: dst laid out [g][lane][16 halves],
// with i/f/o gate rows pre-scaled by 0.5 (exact in f16).
// A blob k order used here (contiguous, matches packB): lanes 0-15 hold row M=L, K=0..15;
// lanes 16-31 hold row M=L-16, K=16..31.
__device__ __forceinline__ void stageA(_Float16* dst, const float* __restrict__ W,
                                       int tid) {
  for (int idx = tid; idx < 256; idx += 128) {
    int g = idx >> 5, L = idx & 31;
    int M  = 16 * g + (L & 15);
    int kb = (L & 16) ? 8 : 0;
    float sc = gate_scale(M);
    _Float16* d = dst + (size_t)(g * 32 + L) * 16;
    const float* src = W + M * 32;
#pragma unroll
    for (int j = 0; j < 16; ++j) {
      int k = 2 * kb + j;
      d[j] = (_Float16)(sc * src[k]);
    }
  }
}

// ---------------- kernel ----------------
// One wave = one 16-row batch tile, 4 waves per block, B/64 blocks.
// Transposed formulation: gates^T[128,16] = W[128,32] @ h^T[32,16] + bias,
// so the recurrent h never needs a cross-lane transpose beyond shfl_xor(16).
// Whole x tile + swizzled weights live in LDS -> zero HBM access inside
// the serial recurrence.

#define WBLK(m, g) (*(const v16h*)&wA[(m)][(g)][lane][0])

__global__ void __launch_bounds__(128) lstm2_wmma(
    const float* __restrict__ x,
    const float* __restrict__ Wih0, const float* __restrict__ Whh0,
    const float* __restrict__ bih0, const float* __restrict__ bhh0,
    const float* __restrict__ Wih1, const float* __restrict__ Whh1,
    const float* __restrict__ bih1, const float* __restrict__ bhh1,
    const float* __restrict__ Wlin, const float* __restrict__ blin,
    float* __restrict__ out) {
  // m=0: W_hh0 ; m=1: W_ih1 ; m=2: W_hh1   (A-format blobs, 8KB each)
  __shared__ __align__(16) _Float16 wA[3][8][32][16];
  __shared__ __align__(16) float xS[SEQT][XPITCH];  // x tile, [t][row] (133KB)
  __shared__ __align__(16) float b0cS[128];   // (b_ih0 + b_hh0) * gate_scale
  __shared__ __align__(16) float w0S[128];    // W_ih0[:,0] * gate_scale
  __shared__ __align__(16) float b1cS[128];   // (b_ih1 + b_hh1) * gate_scale
  __shared__ __align__(16) float wlinS[32];   // W_lin[0,:]

  const int tid = threadIdx.x;

  // ---- one-time staging (coalesced HBM reads; LDS writes 2-way max) ----
  stageA(&wA[0][0][0][0], Whh0, tid);
  stageA(&wA[1][0][0][0], Wih1, tid);
  stageA(&wA[2][0][0][0], Whh1, tid);
  {
    float sc = gate_scale(tid);
    b0cS[tid] = sc * (bih0[tid] + bhh0[tid]);
    w0S[tid]  = sc * Wih0[tid];
    b1cS[tid] = sc * (bih1[tid] + bhh1[tid]);
    if (tid < 32) wlinS[tid] = Wlin[tid];
  }
  {
    const float* xg = x + (size_t)blockIdx.x * 64 * SEQT;  // [64 rows][512 t]
    for (int n4 = tid * 4; n4 < 64 * SEQT; n4 += 128 * 4) {
      v4f v = *(const v4f*)(xg + n4);
      int row = n4 >> 9;        // n4 / 512
      int t   = n4 & 511;
#pragma unroll
      for (int j = 0; j < 4; ++j) xS[t + j][row] = v[j];
    }
  }
  __syncthreads();

  const int  wave = tid >> 5;
  const int  lane = tid & 31;
  const bool lo   = lane < 16;
  const int  hoff = lo ? 0 : 2;          // +8 gate rows for upper half-wave
  const int  tb   = blockIdx.x * 64 + wave * 16;
  const int  col  = lane & 15;           // batch column within tile

  const v4f* B0 = (const v4f*)b0cS;
  const v4f* W0 = (const v4f*)w0S;
  const v4f* B1 = (const v4f*)b1cS;
  const v4f* WL = (const v4f*)wlinS;
  const float* xc = &xS[0][wave * 16 + col];  // advances by 4*XPITCH per iter

  // cell/hidden state in D-layout (lane=batch col, VGPR r = hidden row
  // 16*j + r + 8*(lane>=16)); h also kept in B-operand f16 form.
  v8f  c0a = zero8(), c0b = zero8(), c1a = zero8(), c1b = zero8();
  v8f  h1fa = zero8(), h1fb = zero8();
  v16h h0B = zeroh(), h1B = zeroh();

  for (int t4 = 0; t4 < SEQT; t4 += 4) {
#pragma unroll
    for (int u = 0; u < 4; ++u) {
      const float xv = xc[u * XPITCH];   // ds_load_b32, imm offsets

      // ---------------- layer 0 : gates = b0 + x*w0 + Whh0 @ h0^T
      v8f hfa, hfb;
#pragma unroll
      for (int j = 0; j < 2; ++j) {          // j: hidden 0..15 / 16..31
        v8f gi = fma8s(ld8(W0, 0 + j, hoff), xv, ld8(B0, 0 + j, hoff));
        v8f gf = fma8s(ld8(W0, 2 + j, hoff), xv, ld8(B0, 2 + j, hoff));
        v8f gg = fma8s(ld8(W0, 4 + j, hoff), xv, ld8(B0, 4 + j, hoff));
        v8f go = fma8s(ld8(W0, 6 + j, hoff), xv, ld8(B0, 6 + j, hoff));
        gi = wmma16(WBLK(0, 0 + j), h0B, gi);
        gf = wmma16(WBLK(0, 2 + j), h0B, gf);
        gg = wmma16(WBLK(0, 4 + j), h0B, gg);
        go = wmma16(WBLK(0, 6 + j), h0B, go);
        v8f iv = sig8p(gi), fv = sig8p(gf), gv = tanh8(gg), ov = sig8p(go);
        v8f& c = j ? c0b : c0a;
        c = fv * c + iv * gv;
        (j ? hfb : hfa) = ov * tanh8(c);
      }
      h0B = packB(hfa, hfb, lo);

      // ---------------- layer 1 : gates = b1 + Wih1 @ h0^T + Whh1 @ h1^T
#pragma unroll
      for (int j = 0; j < 2; ++j) {
        v8f gi = ld8(B1, 0 + j, hoff);
        v8f gf = ld8(B1, 2 + j, hoff);
        v8f gg = ld8(B1, 4 + j, hoff);
        v8f go = ld8(B1, 6 + j, hoff);
        gi = wmma16(WBLK(1, 0 + j), h0B, gi);
        gf = wmma16(WBLK(1, 2 + j), h0B, gf);
        gg = wmma16(WBLK(1, 4 + j), h0B, gg);
        go = wmma16(WBLK(1, 6 + j), h0B, go);
        gi = wmma16(WBLK(2, 0 + j), h1B, gi);
        gf = wmma16(WBLK(2, 2 + j), h1B, gf);
        gg = wmma16(WBLK(2, 4 + j), h1B, gg);
        go = wmma16(WBLK(2, 6 + j), h1B, go);
        v8f iv = sig8p(gi), fv = sig8p(gf), gv = tanh8(gg), ov = sig8p(go);
        v8f& c = j ? c1b : c1a;
        c = fv * c + iv * gv;
        (j ? h1fb : h1fa) = ov * tanh8(c);
      }
      h1B = packB(h1fa, h1fb, lo);
    }
    xc += 4 * XPITCH;
  }

  // ---------------- linear head: out[b] = dot(h1_last[:,b], wlin) + blin
  v8f wl0 = ld8(WL, 0, hoff);
  v8f wl1 = ld8(WL, 1, hoff);
  float part = 0.0f;
#pragma unroll
  for (int r = 0; r < 8; ++r)
    part += h1fa[r] * wl0[r] + h1fb[r] * wl1[r];
  part += __shfl_xor(part, 16, 32);       // combine the two half-wave rows
  if (lo) out[tb + lane] = part + blin[0];
}

// ---------------- host launch ----------------

extern "C" void kernel_launch(void* const* d_in, const int* in_sizes, int n_in,
                              void* d_out, int out_size, void* d_ws,
                              size_t ws_size, hipStream_t stream) {
  (void)in_sizes; (void)n_in; (void)d_ws; (void)ws_size; (void)out_size;
  const float* x    = (const float*)d_in[0];
  const float* Wih0 = (const float*)d_in[1];
  const float* Whh0 = (const float*)d_in[2];
  const float* bih0 = (const float*)d_in[3];
  const float* bhh0 = (const float*)d_in[4];
  const float* Wih1 = (const float*)d_in[5];
  const float* Whh1 = (const float*)d_in[6];
  const float* bih1 = (const float*)d_in[7];
  const float* bhh1 = (const float*)d_in[8];
  const float* Wlin = (const float*)d_in[9];
  const float* blin = (const float*)d_in[10];
  float* out = (float*)d_out;

  dim3 grid(BATCH / 64);   // 64 batch rows per block (4 waves * 16)
  dim3 block(128);
  hipLaunchKernelGGL(lstm2_wmma, grid, block, 0, stream,
                     x, Wih0, Whh0, bih0, bhh0,
                     Wih1, Whh1, bih1, bhh1, Wlin, blin, out);
}
